// CTMHead_70677981823276
// MI455X (gfx1250) — matrix operand order repaired and checked by reading, not compiled, not run on hardware
//
#include <hip/hip_runtime.h>
#include <math.h>

// ---------------- CDNA5 WMMA types ----------------
typedef __attribute__((ext_vector_type(16))) _Float16 v16h;
typedef __attribute__((ext_vector_type(8)))  _Float16 v8h;
typedef __attribute__((ext_vector_type(8)))  float    v8f;

#define B_  32
#define H_  1024
#define N_  1024
#define L_  32
#define V_  151936
#define NH_ 64
#define DIN_ 2048
#define TICKS_ 10

// ---------------- device helpers ----------------
__device__ __forceinline__ float gelu_erf(float x) {
    return 0.5f * x * (1.0f + erff(x * 0.70710678118654752f));
}

// A fragment: 16x32 f16, contiguous K. lane<16: K = 0..15; lane>=16: K = 16..31
__device__ __forceinline__ v16h load_a_frag(const _Float16* __restrict__ A, int lda,
                                            int m0, int k0, int lane) {
    int m  = m0 + (lane & 15);
    int kb = (lane >> 4) * 16;
    const _Float16* p = A + (long long)m * lda + k0 + kb;
    v8h lo = *(const v8h*)(p);
    v8h hi = *(const v8h*)(p + 8);
    v16h a;
#pragma unroll
    for (int i = 0; i < 8; ++i) { a[i] = lo[i]; a[i + 8] = hi[i]; }
    return a;
}

// B fragment: 32x16 f16. lane = column n; lane<16: K=0..15, lane>=16: K=16..31 (contiguous in W row-major NxK)
__device__ __forceinline__ v16h load_b_frag(const _Float16* __restrict__ W, int ldw,
                                            int n0, int k0, int lane) {
    int n  = n0 + (lane & 15);
    int kb = (lane >> 4) * 16;
    return *(const v16h*)(W + (long long)n * ldw + k0 + kb);
}

// ---------------- reductions / quantization ----------------
__global__ void k_zero(float* p, int n) {
    int i = blockIdx.x * blockDim.x + threadIdx.x;
    if (i < n) p[i] = 0.0f;
}

__global__ void k_abs_part(const float* __restrict__ W, long long n, float* __restrict__ partials) {
    __shared__ float red[256];
    long long i = (long long)blockIdx.x * blockDim.x + threadIdx.x;
    long long stride = (long long)gridDim.x * blockDim.x;
    float s = 0.0f;
    for (; i < n; i += stride) s += fabsf(W[i]);
    red[threadIdx.x] = s; __syncthreads();
    for (int o = 128; o > 0; o >>= 1) {
        if (threadIdx.x < o) red[threadIdx.x] += red[threadIdx.x + o];
        __syncthreads();
    }
    if (threadIdx.x == 0) partials[blockIdx.x] = red[0];
}

__global__ void k_abs_fin(const float* __restrict__ partials, float* gsum, int idx) {
    __shared__ float red[256];
    int t = threadIdx.x;
    float s = 0.0f;
    for (int i = t; i < 1024; i += 256) s += partials[i];
    red[t] = s; __syncthreads();
    for (int o = 128; o > 0; o >>= 1) { if (t < o) red[t] += red[t + o]; __syncthreads(); }
    if (t == 0) gsum[idx] = red[0];
}

__global__ void k_quantize(const float* __restrict__ W, _Float16* __restrict__ out,
                           long long n, const float* __restrict__ gsum, int idx, float invCount) {
    long long i = (long long)blockIdx.x * blockDim.x + threadIdx.x;
    if (i >= n) return;
    float gamma = gsum[idx] * invCount + 1e-8f;
    float q = rintf(W[i] / gamma);
    q = fminf(1.0f, fmaxf(-1.0f, q));
    out[i] = (_Float16)q;
}

// one wave per row: sum of quantized Wo row (rank-1 logits trick)
__global__ void k_rowsum_wo(const float* __restrict__ Wo, const float* __restrict__ gsum,
                            float* __restrict__ rowsum) {
    int row = blockIdx.x * (blockDim.x >> 5) + (threadIdx.x >> 5);
    if (row >= V_) return;
    int lane = threadIdx.x & 31;
    float gamma = gsum[10] / ((float)V_ * (float)H_) + 1e-8f;
    const float* p = Wo + (long long)row * H_;
    float s = 0.0f;
    for (int j = lane; j < H_; j += 32) {
        float q = rintf(p[j] / gamma);
        s += fminf(1.0f, fmaxf(-1.0f, q));
    }
    for (int o = 16; o > 0; o >>= 1) s += __shfl_down(s, o, 32);
    if (lane == 0) rowsum[row] = s;
}

// single block: gamma(Wc) then sum of quantized Wc -> gsum[12]
__global__ void k_wc_sum(const float* __restrict__ Wc, float* gsum) {
    __shared__ float red[256];
    __shared__ float gam;
    int t = threadIdx.x;
    float s = 0.0f;
    for (int i = t; i < H_; i += 256) s += fabsf(Wc[i]);
    red[t] = s; __syncthreads();
    for (int o = 128; o > 0; o >>= 1) { if (t < o) red[t] += red[t + o]; __syncthreads(); }
    if (t == 0) gam = red[0] / (float)H_ + 1e-8f;
    __syncthreads();
    float g = gam, q = 0.0f;
    for (int i = t; i < H_; i += 256) {
        float v = rintf(Wc[i] / g);
        q += fminf(1.0f, fmaxf(-1.0f, v));
    }
    __syncthreads();
    red[t] = q; __syncthreads();
    for (int o = 128; o > 0; o >>= 1) { if (t < o) red[t] += red[t + o]; __syncthreads(); }
    if (t == 0) gsum[12] = red[0];
}

__global__ void k_cvt(const float* __restrict__ src, _Float16* __restrict__ dst,
                      long long n, float scale) {
    long long i = (long long)blockIdx.x * blockDim.x + threadIdx.x;
    if (i < n) dst[i] = (_Float16)(src[i] * scale);
}

// ---------------- state init ----------------
__global__ void k_catin(const float* __restrict__ l0, const float* __restrict__ l1,
                        _Float16* __restrict__ xin) {
    int i = blockIdx.x * blockDim.x + threadIdx.x;     // b*2048 + j
    if (i >= B_ * DIN_) return;
    int b = i >> 11, j = i & 2047;
    float v = (j < H_) ? l0[b * H_ + j] : l1[b * H_ + j - H_];
    xin[i] = (_Float16)v;
}

__global__ void k_init_state(const float* __restrict__ iz, const float* __restrict__ ih,
                             _Float16* __restrict__ xcat, _Float16* __restrict__ hist) {
    int i = blockIdx.x * blockDim.x + threadIdx.x;     // b*1024 + n
    if (i >= B_ * N_) return;
    int b = i >> 10, n = i & 1023;
    xcat[(long long)b * DIN_ + n] = (_Float16)iz[n];
#pragma unroll 4
    for (int l = 0; l < L_; ++l) hist[(long long)i * L_ + l] = (_Float16)ih[n * L_ + l];
}

// ---------------- generic WMMA GEMM: Y = act(X @ Wq^T + bias) ----------------
// A: MxK f16 (lda); W: NxK f16; outH f16 at [m*ldo + ocol + n]; optional f32 out [m*N + n]
__global__ void k_gemm(const _Float16* __restrict__ A, int lda,
                       const _Float16* __restrict__ W,
                       const float* __restrict__ bias,
                       _Float16* __restrict__ outH, int ldo, int ocol,
                       float* __restrict__ outF,
                       int M, int N, int K, int doGelu) {
    int lane = threadIdx.x & 31;
    int wv   = threadIdx.x >> 5;
    int nt   = (blockIdx.x * (blockDim.x >> 5) + wv) * 16;
    int m0   = blockIdx.y * 16;
    if (nt >= N || m0 >= M) return;

    v8f acc = {};
    for (int k0 = 0; k0 < K; k0 += 32) {
        v16h a = load_a_frag(A, lda, m0, k0, lane);
        v16h b = load_b_frag(W, K, nt, k0, lane);
        acc = __builtin_amdgcn_wmma_f32_16x16x32_f16(false, a, false, b,
                                                     (short)0, acc, false, false);
    }
    int nl = lane & 15;
    int mh = (lane >> 4) * 8;
#pragma unroll
    for (int r = 0; r < 8; ++r) {
        int m = m0 + mh + r;
        int n = nt + nl;
        float y = acc[r] + bias[n];
        if (doGelu) y = gelu_erf(y);
        outH[(long long)m * ldo + ocol + n] = (_Float16)y;
        if (outF) outF[(long long)m * N + n] = y;
    }
}

// ---------------- history write (circular slot) ----------------
__global__ void k_histw(const _Float16* __restrict__ pre, _Float16* __restrict__ hist, int slot) {
    int i = blockIdx.x * blockDim.x + threadIdx.x;     // b*1024 + n
    if (i >= B_ * N_) return;
    hist[(long long)i * L_ + slot] = pre[i];
}

// ---------------- neuron model: per-neuron (32x32)@(32x64) via WMMA ----------------
// hist (B,N,L) f16 circular with offset 'off'; W1h (N,NH,L) f16; h = gelu(. + b1) -> f32 (B,N,NH)
__global__ void k_neuron(const _Float16* __restrict__ hist, const _Float16* __restrict__ W1h,
                         const float* __restrict__ b1, float* __restrict__ hout, int off) {
    int n    = blockIdx.x;              // neuron id
    int wv   = threadIdx.x >> 5;        // 0..7
    int lane = threadIdx.x & 31;
    int mt   = (wv & 1) * 16;           // batch tile
    int nt   = (wv >> 1) * 16;          // NH tile

    // A fragment: A[m][k] = hist[(m*N + n)*L + (off+k)&31]
    int m  = mt + (lane & 15);
    int kb = (lane >> 4) * 16;
    const _Float16* hrow = hist + ((long long)m * N_ + n) * L_;
    v16h a;
#pragma unroll
    for (int j = 0; j < 8; ++j) {
        a[j]     = hrow[(off + kb + j) & 31];
        a[j + 8] = hrow[(off + kb + 8 + j) & 31];
    }
    // B fragment: B[l][h] = W1h[(n*NH + h)*L + l] (contiguous in l)
    int hcol = nt + (lane & 15);
    int kb2  = (lane >> 4) * 16;
    v16h b = *(const v16h*)(W1h + ((long long)n * NH_ + hcol) * L_ + kb2);

    v8f acc = {};
    acc = __builtin_amdgcn_wmma_f32_16x16x32_f16(false, a, false, b,
                                                 (short)0, acc, false, false);
    int nl = lane & 15;
    int mh = (lane >> 4) * 8;
#pragma unroll
    for (int r = 0; r < 8; ++r) {
        int mb = mt + mh + r;
        int hh = nt + nl;
        float y = gelu_erf(acc[r] + b1[n * NH_ + hh]);
        hout[((long long)mb * N_ + n) * NH_ + hh] = y;
    }
}

// ---------------- z = h @ W2 (per neuron) + b2 ; update postsum & xcat ----------------
__global__ void k_w2z(const float* __restrict__ hout, const float* __restrict__ W2,
                      const float* __restrict__ b2, float* __restrict__ z,
                      float* __restrict__ postsum, _Float16* __restrict__ xcat) {
    int i = blockIdx.x * blockDim.x + threadIdx.x;     // b*1024 + n
    if (i >= B_ * N_) return;
    int n = i & 1023;
    const float* hp = hout + (long long)i * NH_;
    const float* wp = W2 + n * NH_;
    float s = b2[n];
#pragma unroll 8
    for (int h = 0; h < NH_; ++h) s += hp[h] * wp[h];
    z[i] = s;
    postsum[i] += s;
    int b = i >> 10;
    xcat[(long long)b * DIN_ + n] = (_Float16)s;
}

// ---------------- sync = cos(q,k) ----------------
__global__ void k_sync(const float* __restrict__ q, const float* __restrict__ k,
                       float* __restrict__ outSync) {
    __shared__ float rq[256], rk[256], rqk[256];
    int b = blockIdx.x, t = threadIdx.x;
    float qv = q[b * 256 + t], kv = k[b * 256 + t];
    rq[t] = qv * qv; rk[t] = kv * kv; rqk[t] = qv * kv; __syncthreads();
    for (int o = 128; o > 0; o >>= 1) {
        if (t < o) { rq[t] += rq[t + o]; rk[t] += rk[t + o]; rqk[t] += rqk[t + o]; }
        __syncthreads();
    }
    if (t == 0) {
        float nq = fmaxf(sqrtf(rq[0]), 1e-8f);
        float nk = fmaxf(sqrtf(rk[0]), 1e-8f);
        outSync[b] = rqk[0] / (nq * nk);
    }
}

// ---------------- mz = mean(z); certainty = sigmoid(mz*sumqWc + bc) ----------------
__global__ void k_mzcert(const float* __restrict__ z, const float* __restrict__ gsum,
                         const float* __restrict__ bc, float* __restrict__ mz,
                         float* __restrict__ outCert) {
    __shared__ float red[256];
    int b = blockIdx.x, t = threadIdx.x;
    float s = 0.0f;
    for (int n = t; n < N_; n += 256) s += z[b * N_ + n];
    red[t] = s; __syncthreads();
    for (int o = 128; o > 0; o >>= 1) { if (t < o) red[t] += red[t + o]; __syncthreads(); }
    if (t == 0) {
        float m = red[0] / (float)N_;
        mz[b] = m;
        float c = m * gsum[12] + bc[0];
        outCert[b] = 1.0f / (1.0f + expf(-c));
    }
}

// ---------------- logits[b,v] = mz[b]*rowsum[v] + bo[v] ----------------
__global__ void k_logits(const float* __restrict__ mz, const float* __restrict__ rowsum,
                         const float* __restrict__ bo, float* __restrict__ out) {
    long long i = (long long)blockIdx.x * blockDim.x + threadIdx.x;
    if (i >= (long long)B_ * V_) return;
    int b = (int)(i / V_);
    int v = (int)(i - (long long)b * V_);
    out[i] = mz[b] * rowsum[v] + bo[v];
}

// ================= host side =================
static inline size_t alignUp(size_t x) { return (x + 255) & ~(size_t)255; }

static void gemm(hipStream_t s, const _Float16* A, int lda, const _Float16* W,
                 const float* bias, _Float16* outH, int ldo, int ocol, float* outF,
                 int M, int N, int K, int act) {
    dim3 grid((N / 16 + 3) / 4, M / 16);
    k_gemm<<<grid, 128, 0, s>>>(A, lda, W, bias, outH, ldo, ocol, outF, M, N, K, act);
}

static void quantW(hipStream_t s, const float* W, long long n, float* partials,
                   float* gsum, int idx, _Float16* out) {
    k_abs_part<<<1024, 256, 0, s>>>(W, n, partials);
    k_abs_fin<<<1, 256, 0, s>>>(partials, gsum, idx);
    if (out) {
        int blocks = (int)((n + 255) / 256);
        k_quantize<<<blocks, 256, 0, s>>>(W, out, n, gsum, idx, (float)(1.0 / (double)n));
    }
}

extern "C" void kernel_launch(void* const* d_in, const int* in_sizes, int n_in,
                              void* d_out, int out_size, void* d_ws, size_t ws_size,
                              hipStream_t stream) {
    (void)in_sizes; (void)n_in; (void)out_size; (void)ws_size;
    const float* l0   = (const float*)d_in[0];
    const float* l1   = (const float*)d_in[1];
    const float* iz   = (const float*)d_in[2];
    const float* ih   = (const float*)d_in[3];
    const float* Wp   = (const float*)d_in[4];   const float* bp  = (const float*)d_in[5];
    const float* We0  = (const float*)d_in[6];   const float* be0 = (const float*)d_in[7];
    const float* We1  = (const float*)d_in[8];   const float* be1 = (const float*)d_in[9];
    const float* We2  = (const float*)d_in[10];  const float* be2 = (const float*)d_in[11];
    const float* Wb   = (const float*)d_in[12];  const float* bb  = (const float*)d_in[13];
    const float* Wd0  = (const float*)d_in[14];  const float* bd0 = (const float*)d_in[15];
    const float* Wd1  = (const float*)d_in[16];  const float* bd1 = (const float*)d_in[17];
    const float* Wd2  = (const float*)d_in[18];  const float* bd2 = (const float*)d_in[19];
    const float* W1   = (const float*)d_in[20];  const float* b1  = (const float*)d_in[21];
    const float* W2   = (const float*)d_in[22];  const float* b2  = (const float*)d_in[23];
    const float* Wo   = (const float*)d_in[24];  const float* bo  = (const float*)d_in[25];
    const float* Wc   = (const float*)d_in[26];  const float* bc  = (const float*)d_in[27];
    const float* Wq   = (const float*)d_in[28];  const float* bq  = (const float*)d_in[29];
    const float* Wk   = (const float*)d_in[30];  const float* bk  = (const float*)d_in[31];

    // ---- workspace carve-up ----
    char* base = (char*)d_ws;
    size_t cur = 0;
    auto take = [&](size_t bytes) { void* p = base + cur; cur = alignUp(cur + bytes); return p; };

    float*    partials = (float*)take(1024 * sizeof(float));
    float*    gsum     = (float*)take(16 * sizeof(float));
    _Float16* qWp  = (_Float16*)take((size_t)H_ * DIN_ * 2);
    _Float16* qWe0 = (_Float16*)take((size_t)1024 * 2048 * 2);
    _Float16* qWe1 = (_Float16*)take((size_t)512 * 1024 * 2);
    _Float16* qWe2 = (_Float16*)take((size_t)256 * 512 * 2);
    _Float16* qWb  = (_Float16*)take((size_t)256 * 256 * 2);
    _Float16* qWd0 = (_Float16*)take((size_t)512 * 256 * 2);
    _Float16* qWd1 = (_Float16*)take((size_t)1024 * 512 * 2);
    _Float16* qWd2 = (_Float16*)take((size_t)1024 * 1024 * 2);
    _Float16* qWq  = (_Float16*)take((size_t)256 * 1024 * 2);
    _Float16* qWk  = (_Float16*)take((size_t)256 * 1024 * 2);
    _Float16* W1h  = (_Float16*)take((size_t)N_ * NH_ * L_ * 2);
    float*    rowsum = (float*)take((size_t)V_ * sizeof(float));
    _Float16* xin  = (_Float16*)take((size_t)B_ * DIN_ * 2);
    _Float16* xcat = (_Float16*)take((size_t)B_ * DIN_ * 2);
    _Float16* actA = (_Float16*)take((size_t)B_ * 1024 * 2);
    _Float16* actB = (_Float16*)take((size_t)B_ * 1024 * 2);
    _Float16* hist = (_Float16*)take((size_t)B_ * N_ * L_ * 2);
    float*    hout = (float*)take((size_t)B_ * N_ * NH_ * sizeof(float));
    float*    zbuf = (float*)take((size_t)B_ * N_ * sizeof(float));
    float*    psum = (float*)take((size_t)B_ * N_ * sizeof(float));
    _Float16* mph  = (_Float16*)take((size_t)B_ * N_ * 2);
    float*    qf   = (float*)take((size_t)B_ * 256 * sizeof(float));
    float*    kf   = (float*)take((size_t)B_ * 256 * sizeof(float));
    _Float16* qkh  = (_Float16*)take((size_t)B_ * 256 * 2);
    float*    mz   = (float*)take(B_ * sizeof(float));

    float* outLogits = (float*)d_out;
    float* outSync   = outLogits + (long long)B_ * V_;
    float* outCert   = outSync + B_;

    // ---- 0) clear accumulators (must happen every call: graph replays) ----
    k_zero<<<(B_ * N_ + 255) / 256, 256, 0, stream>>>(psum, B_ * N_);

    // ---- 1) BitNet absmean quantization (deterministic two-stage) ----
    quantW(stream, Wp,  (long long)H_ * DIN_,    partials, gsum, 0, qWp);
    quantW(stream, We0, (long long)1024 * 2048,  partials, gsum, 1, qWe0);
    quantW(stream, We1, (long long)512 * 1024,   partials, gsum, 2, qWe1);
    quantW(stream, We2, (long long)256 * 512,    partials, gsum, 3, qWe2);
    quantW(stream, Wb,  (long long)256 * 256,    partials, gsum, 4, qWb);
    quantW(stream, Wd0, (long long)512 * 256,    partials, gsum, 5, qWd0);
    quantW(stream, Wd1, (long long)1024 * 512,   partials, gsum, 6, qWd1);
    quantW(stream, Wd2, (long long)1024 * 1024,  partials, gsum, 7, qWd2);
    quantW(stream, Wq,  (long long)256 * 1024,   partials, gsum, 8, qWq);
    quantW(stream, Wk,  (long long)256 * 1024,   partials, gsum, 9, qWk);
    quantW(stream, Wo,  (long long)V_ * H_,      partials, gsum, 10, nullptr);
    k_rowsum_wo<<<(V_ + 7) / 8, 256, 0, stream>>>(Wo, gsum, rowsum);   // rank-1 logits trick
    k_wc_sum<<<1, 256, 0, stream>>>(Wc, gsum);

    // ---- 2) plain f32->f16 conversion of W1 (not quantized in reference) ----
    k_cvt<<<(int)(((long long)N_ * NH_ * L_ + 255) / 256), 256, 0, stream>>>(
        W1, W1h, (long long)N_ * NH_ * L_, 1.0f);

    // ---- 3) initial state ----
    k_catin<<<(B_ * DIN_ + 255) / 256, 256, 0, stream>>>(l0, l1, xin);
    k_init_state<<<(B_ * N_ + 255) / 256, 256, 0, stream>>>(iz, ih, xcat, hist);

    // ---- 4) attn = bitlin(cat(l0,l1), Wp) -> write f16 into xcat[:,1024:2048] ----
    gemm(stream, xin, DIN_, qWp, bp, xcat, DIN_, H_ /*ocol*/, nullptr, B_, H_, DIN_, 0);

    // ---- 5) tick loop ----
    for (int t = 0; t < TICKS_; ++t) {
        // synapse U-Net (all outputs gelu'd, f16, ping-pong buffers)
        gemm(stream, xcat, DIN_, qWe0, be0, actA, 1024, 0, nullptr, B_, 1024, 2048, 1);
        gemm(stream, actA, 1024, qWe1, be1, actB,  512, 0, nullptr, B_,  512, 1024, 1);
        gemm(stream, actB,  512, qWe2, be2, actA,  256, 0, nullptr, B_,  256,  512, 1);
        gemm(stream, actA,  256, qWb,  bb,  actB,  256, 0, nullptr, B_,  256,  256, 1);
        gemm(stream, actB,  256, qWd0, bd0, actA,  512, 0, nullptr, B_,  512,  256, 1);
        gemm(stream, actA,  512, qWd1, bd1, actB, 1024, 0, nullptr, B_, 1024,  512, 1);
        gemm(stream, actB, 1024, qWd2, bd2, actA, 1024, 0, nullptr, B_, 1024, 1024, 1); // pre
        // circular history: write pre into slot t; effective order offset = t+1
        k_histw<<<(B_ * N_ + 255) / 256, 256, 0, stream>>>(actA, hist, t & 31);
        // neuron-level model: 1024 tiny WMMA GEMMs (one block / neuron, 8 waves)
        k_neuron<<<N_, 256, 0, stream>>>(hist, W1h, b1, hout, (t + 1) & 31);
        // z = h @ W2 + b2; accumulate postsum; feed z back into xcat (f16)
        k_w2z<<<(B_ * N_ + 255) / 256, 256, 0, stream>>>(hout, W2, b2, zbuf, psum, xcat);
    }

    // ---- 6) mean_post, q/k bitlin, cosine sync ----
    k_cvt<<<(B_ * N_ + 255) / 256, 256, 0, stream>>>(psum, mph, (long long)B_ * N_,
                                                     1.0f / (float)TICKS_);
    gemm(stream, mph, N_, qWq, bq, qkh, 256, 0, qf, B_, 256, N_, 0);
    gemm(stream, mph, N_, qWk, bk, qkh, 256, 0, kf, B_, 256, N_, 0);
    k_sync<<<B_, 256, 0, stream>>>(qf, kf, outSync);

    // ---- 7) rank-1 logits + certainty ----
    k_mzcert<<<B_, 256, 0, stream>>>(zbuf, gsum, bc, mz, outCert);
    {
        long long total = (long long)B_ * V_;
        k_logits<<<(int)((total + 255) / 256), 256, 0, stream>>>(mz, rowsum, bo, outLogits);
    }
}
